// MultiheadAttention_20684562498233
// MI455X (gfx1250) — matrix operand order repaired and checked by reading, not compiled, not run on hardware
//
#include <hip/hip_runtime.h>

// ---------------------------------------------------------------------------
// MI455X (gfx1250) multi-head attention, bf16 WMMA path (wave32).
// D = A x B + C via v_wmma_f32_16x16x32_bf16; all operand fragments are
// two contiguous 16-byte loads per the ISA 7.12.2 lane layouts.
// ---------------------------------------------------------------------------

typedef __attribute__((ext_vector_type(16))) __bf16       v16bf;
typedef __attribute__((ext_vector_type(8)))  float        v8f;
typedef __attribute__((ext_vector_type(4)))  float        v4f;
typedef __attribute__((ext_vector_type(4)))  unsigned int v4u;
typedef __attribute__((ext_vector_type(2)))  unsigned int v2u;

#define DEV __device__ __forceinline__

#define B_SZ  2
#define S_SZ  2048
#define D_SZ  1024
#define H_SZ  16
#define DH_SZ 64
#define M_ALL (B_SZ * S_SZ)   // 4096
#define LDP   40              // LDS row stride (elems): 80B -> 16B-aligned rows

// float -> bf16, round-to-nearest-even
DEV unsigned short f2bf(float f) {
  unsigned u = __builtin_bit_cast(unsigned, f);
  u += 0x7fffu + ((u >> 16) & 1u);
  return (unsigned short)(u >> 16);
}

DEV v8f wmma_bf16(v16bf a, v16bf b, v8f c) {
  // (neg_a, A, neg_b, B, c_mod, C, reuse_a, reuse_b)
  return __builtin_amdgcn_wmma_f32_16x16x32_bf16(false, a, false, b,
                                                 (short)0, c, false, false);
}

struct frag_bits { v4u lo, hi; };

DEV v16bf load_frag16(const unsigned short* p0, const unsigned short* p1) {
  frag_bits t;
  t.lo = *(const v4u*)p0;
  t.hi = *(const v4u*)p1;
  return __builtin_bit_cast(v16bf, t);
}

// A-operand (16x32 MxK): lane&15 = row; VGPR0-3: K={0..7}+half*8,
// VGPR4-7: K={16..23}+half*8. rowptr = this lane's row, K contiguous.
DEV v16bf load_a_frag(const unsigned short* rowptr, int koff, int half) {
  const unsigned short* p = rowptr + koff + (half << 3);
  return load_frag16(p, p + 16);
}

// B-operand (32x16 KxN): lane&15 = column; VGPR0-3: K={0..7}+half*8,
// VGPR4-7: K={16..23}+half*8 (same K order as A).
DEV v16bf load_b_frag(const unsigned short* colptr, int koff, int half) {
  const unsigned short* p = colptr + koff + (half << 3);
  return load_frag16(p, p + 16);
}

// ---------------------------------------------------------------------------
// Staging: A tile 128x32 -> bf16 LDS (row-major), W tile 32x64 -> LDS
// transposed [n][k] so B fragments are K-contiguous. All-vector loads.
// ---------------------------------------------------------------------------
template <int ABF16>
DEV void stage_tile(unsigned short (*__restrict__ As)[LDP],
                    unsigned short (*__restrict__ Ws)[LDP],
                    const void* __restrict__ Ain, const float* __restrict__ Wg,
                    int m0, int n0, int k0, int tid) {
  if constexpr (ABF16) {
#pragma unroll
    for (int it = 0; it < 2; ++it) {          // 512 x 16B chunks
      int i = tid + it * 256;
      int r = i >> 2, c8 = (i & 3) * 8;
      const v4u* src = (const v4u*)((const unsigned short*)Ain +
                                    (size_t)(m0 + r) * D_SZ + k0 + c8);
      *(v4u*)&As[r][c8] = *src;
    }
  } else {
#pragma unroll
    for (int it = 0; it < 4; ++it) {          // 1024 x float4 chunks
      int i = tid + it * 256;
      int r = i >> 3, c4 = (i & 7) * 4;
      v4f f = *(const v4f*)((const float*)Ain +
                            (size_t)(m0 + r) * D_SZ + k0 + c4);
      v2u p;
      p[0] = (unsigned)f2bf(f[0]) | ((unsigned)f2bf(f[1]) << 16);
      p[1] = (unsigned)f2bf(f[2]) | ((unsigned)f2bf(f[3]) << 16);
      *(v2u*)&As[r][c4] = p;
    }
  }
#pragma unroll
  for (int it = 0; it < 2; ++it) {            // 512 x float4 chunks of W
    int i = tid + it * 256;
    int kk = i >> 4, n4 = (i & 15) * 4;
    v4f f = *(const v4f*)(Wg + (size_t)(k0 + kk) * D_SZ + n0 + n4);
#pragma unroll
    for (int j = 0; j < 4; ++j) Ws[n4 + j][kk] = f2bf(f[j]);
  }
}

// ---------------------------------------------------------------------------
// GEMM: [4096 x 1024] @ [1024 x 1024], bf16 WMMA, f32 accumulate.
// MODE 0: out bf16 head-split [B,H,S,64]   (Q, K projections)
// MODE 1: out bf16 transposed [B,H,64,S]   (V projection)
// MODE 2: out f32 row-major   [4096,1024]  (final projection)
// 256 thr = 8 waves; block tile 128x64; wave tile 32x32; K-step 32;
// double-buffered LDS.
// ---------------------------------------------------------------------------
template <int ABF16, int MODE>
__global__ __launch_bounds__(256)
void mha_gemm(const void* __restrict__ Ain, const float* __restrict__ Wg,
              unsigned short* __restrict__ outB, float* __restrict__ outF) {
  __shared__ alignas(16) unsigned short AsB[2][128][LDP];
  __shared__ alignas(16) unsigned short WsB[2][64][LDP];

  const int tid  = threadIdx.x;
  const int n0   = blockIdx.x * 64;
  const int m0   = blockIdx.y * 128;
  const int wave = tid >> 5, lane = tid & 31;
  const int half = lane >> 4, l16 = lane & 15;
  const int wm = (wave >> 1) * 32, wn = (wave & 1) * 32;

  v8f acc[2][2];
#pragma unroll
  for (int i = 0; i < 2; ++i)
#pragma unroll
    for (int j = 0; j < 2; ++j)
#pragma unroll
      for (int e = 0; e < 8; ++e) acc[i][j][e] = 0.f;

  stage_tile<ABF16>(AsB[0], WsB[0], Ain, Wg, m0, n0, 0, tid);

  const int KT = D_SZ / 32;
  for (int kt = 0; kt < KT; ++kt) {
    __syncthreads();
    const int cur = kt & 1;
    if (kt + 1 < KT) {
      if (kt + 2 < KT) {  // global_prefetch_b8 two tiles ahead
        const char* base = (const char*)Ain;
        size_t esz = ABF16 ? 2 : 4;
        __builtin_prefetch(
            base + ((size_t)(m0 + (tid & 127)) * D_SZ + (kt + 2) * 32) * esz,
            0, 0);
      }
      stage_tile<ABF16>(AsB[cur ^ 1], WsB[cur ^ 1], Ain, Wg, m0, n0,
                        (kt + 1) * 32, tid);
    }
    v16bf af[2], bfr[2];
#pragma unroll
    for (int i = 0; i < 2; ++i)
      af[i] = load_a_frag(&AsB[cur][wm + 16 * i + l16][0], 0, half);
#pragma unroll
    for (int j = 0; j < 2; ++j)
      bfr[j] = load_b_frag(&WsB[cur][wn + 16 * j + l16][0], 0, half);
#pragma unroll
    for (int i = 0; i < 2; ++i)
#pragma unroll
      for (int j = 0; j < 2; ++j)
        acc[i][j] = wmma_bf16(af[i], bfr[j], acc[i][j]);
  }

  // C/D layout: VGPR e -> row half*8+e, col = lane&15
#pragma unroll
  for (int i = 0; i < 2; ++i)
#pragma unroll
    for (int j = 0; j < 2; ++j)
#pragma unroll
      for (int e = 0; e < 8; ++e) {
        int m = m0 + wm + 16 * i + (half << 3) + e;
        int n = n0 + wn + 16 * j + l16;
        float v = acc[i][j][e];
        if constexpr (MODE == 2) {
          outF[(size_t)m * D_SZ + n] = v;
        } else {
          int b = m >> 11, s = m & (S_SZ - 1), h = n >> 6, d = n & 63;
          if constexpr (MODE == 0)
            outB[(((size_t)b * H_SZ + h) * S_SZ + s) * DH_SZ + d] = f2bf(v);
          else
            outB[(((size_t)b * H_SZ + h) * DH_SZ + d) * S_SZ + s] = f2bf(v);
        }
      }
}

// ---------------------------------------------------------------------------
// Flash attention. 8 waves/block; each wave owns 16 query rows of one (b,h).
// Per 64-key tile: 8 WMMA (scores) + online softmax + 8 WMMA (P*V).
// Mask is faithful to reference: (score==0) && (q>=k) -> -MAX, then 0.
// ---------------------------------------------------------------------------
__global__ __launch_bounds__(256)
void mha_attn(const unsigned short* __restrict__ qp,
              const unsigned short* __restrict__ kp,
              const unsigned short* __restrict__ vt,
              unsigned short* __restrict__ attn) {
  __shared__ alignas(16) unsigned short Ps[8][16][72];  // per-wave P tile

  const int tid  = threadIdx.x;
  const int wave = tid >> 5, lane = tid & 31;
  const int half = lane >> 4, l16 = lane & 15;
  const int b = blockIdx.z, h = blockIdx.y, bh = b * H_SZ + h;
  const int qr0 = blockIdx.x * 128 + wave * 16;

  const unsigned short* qrow = qp + ((size_t)bh * S_SZ + qr0 + l16) * DH_SZ;
  v16bf qf0 = load_a_frag(qrow, 0, half);    // K = 0..31 of d_h
  v16bf qf1 = load_a_frag(qrow, 32, half);   // K = 32..63

  v8f o[4], zero;
  float mrun[8], lrun[8];
#pragma unroll
  for (int e = 0; e < 8; ++e) zero[e] = 0.f;
#pragma unroll
  for (int a = 0; a < 4; ++a) o[a] = zero;
#pragma unroll
  for (int e = 0; e < 8; ++e) { mrun[e] = -3.0e38f; lrun[e] = 0.f; }

  for (int key0 = 0; key0 < S_SZ; key0 += 64) {
    // ---- scores: 4 tiles of 16x16, each = 2 WMMA k-steps over d_h=64 ----
    v8f sc[4];
#pragma unroll
    for (int t = 0; t < 4; ++t) {
      const unsigned short* kcol =
          kp + ((size_t)bh * S_SZ + key0 + 16 * t + l16) * DH_SZ;
      v8f z = wmma_bf16(qf0, load_b_frag(kcol, 0, half), zero);
      sc[t]  = wmma_bf16(qf1, load_b_frag(kcol, 32, half), z);
    }

    // ---- scale + reference mask + row max ----
    float mnew[8];
#pragma unroll
    for (int e = 0; e < 8; ++e) mnew[e] = mrun[e];
#pragma unroll
    for (int t = 0; t < 4; ++t)
#pragma unroll
      for (int e = 0; e < 8; ++e) {
        float s = sc[t][e] * 0.125f;  // 1/sqrt(64)
        int qg = qr0 + (half << 3) + e;
        int kg = key0 + t * 16 + l16;
        if (s == 0.0f && qg >= kg) s = -3.0e38f;
        sc[t][e] = s;
        mnew[e] = fmaxf(mnew[e], s);
      }
#pragma unroll
    for (int off = 1; off < 16; off <<= 1)
#pragma unroll
      for (int e = 0; e < 8; ++e)
        mnew[e] = fmaxf(mnew[e], __shfl_xor(mnew[e], off, 16));

    // ---- online softmax update ----
    float corr[8], rsum[8];
#pragma unroll
    for (int e = 0; e < 8; ++e) {
      corr[e] = __expf(mrun[e] - mnew[e]);
      mrun[e] = mnew[e];
      rsum[e] = 0.f;
    }
#pragma unroll
    for (int t = 0; t < 4; ++t)
#pragma unroll
      for (int e = 0; e < 8; ++e) {
        float p = __expf(sc[t][e] - mnew[e]);  // masked -> exp(-huge) = 0
        sc[t][e] = p;
        rsum[e] += p;
      }
#pragma unroll
    for (int off = 1; off < 16; off <<= 1)
#pragma unroll
      for (int e = 0; e < 8; ++e) rsum[e] += __shfl_xor(rsum[e], off, 16);
#pragma unroll
    for (int e = 0; e < 8; ++e) lrun[e] = lrun[e] * corr[e] + rsum[e];
#pragma unroll
    for (int a = 0; a < 4; ++a)
#pragma unroll
      for (int e = 0; e < 8; ++e) o[a][e] *= corr[e];

    // ---- C-layout -> A-layout via per-wave LDS tile ----
    // Ps[wave] is wave-private; DS ops are in-order within a wave, the
    // fences only prevent compiler reordering.
    __threadfence_block();
#pragma unroll
    for (int t = 0; t < 4; ++t)
#pragma unroll
      for (int e = 0; e < 8; ++e)
        Ps[wave][(half << 3) + e][t * 16 + l16] = f2bf(sc[t][e]);
    __threadfence_block();

    // ---- O += P * V : 2 k-steps x 4 dh tiles ----
#pragma unroll
    for (int kb = 0; kb < 2; ++kb) {
      v16bf pa = load_a_frag(&Ps[wave][l16][0], kb * 32, half);
#pragma unroll
      for (int a = 0; a < 4; ++a) {
        const unsigned short* vcol =
            vt + ((size_t)bh * DH_SZ + a * 16 + l16) * S_SZ + key0;
        o[a] = wmma_bf16(pa, load_b_frag(vcol, kb * 32, half), o[a]);
      }
    }
  }

  // ---- normalize and store in concat-head layout [B,S,H*64] (bf16) ----
#pragma unroll
  for (int e = 0; e < 8; ++e) lrun[e] = 1.f / lrun[e];
#pragma unroll
  for (int a = 0; a < 4; ++a)
#pragma unroll
    for (int e = 0; e < 8; ++e) {
      int srow = qr0 + (half << 3) + e;
      int d = a * 16 + l16;
      attn[((size_t)b * S_SZ + srow) * D_SZ + h * DH_SZ + d] =
          f2bf(o[a][e] * lrun[e]);
    }
}

// ---------------------------------------------------------------------------
extern "C" void kernel_launch(void* const* d_in, const int* in_sizes, int n_in,
                              void* d_out, int out_size, void* d_ws,
                              size_t ws_size, hipStream_t stream) {
  (void)in_sizes; (void)n_in; (void)out_size;
  const float* q   = (const float*)d_in[0];
  const float* k   = (const float*)d_in[1];
  const float* v   = (const float*)d_in[2];
  const float* w_q = (const float*)d_in[3];
  const float* w_k = (const float*)d_in[4];
  const float* w_v = (const float*)d_in[5];
  const float* w_o = (const float*)d_in[6];

  const size_t SEG = (size_t)B_SZ * H_SZ * S_SZ * DH_SZ;  // 4,194,304 elems
  if (ws_size < 4 * SEG * sizeof(unsigned short)) return;
  unsigned short* qp = (unsigned short*)d_ws;
  unsigned short* kp = qp + SEG;
  unsigned short* vt = kp + SEG;
  unsigned short* at = vt + SEG;

  dim3 gg(D_SZ / 64, M_ALL / 128);  // (16, 32)
  mha_gemm<0, 0><<<gg, 256, 0, stream>>>(q, w_q, qp, nullptr);
  mha_gemm<0, 0><<<gg, 256, 0, stream>>>(k, w_k, kp, nullptr);
  mha_gemm<0, 1><<<gg, 256, 0, stream>>>(v, w_v, vt, nullptr);
  mha_attn<<<dim3(S_SZ / 128, H_SZ, B_SZ), 256, 0, stream>>>(qp, kp, vt, at);
  mha_gemm<1, 2><<<gg, 256, 0, stream>>>(at, w_o, nullptr, (float*)d_out);
}
